// BadhanauAttention_46935402611472
// MI455X (gfx1250) — matrix operand order repaired and checked by reading, not compiled, not run on hardware
//
#include <hip/hip_runtime.h>
#include <hip/hip_bf16.h>

#define BB 32
#define SS 2048
#define EE 1024   // ENC (K of the big GEMM)
#define DD 1024   // DEC
#define AA 1024   // ATT (N of the big GEMM)

#define CHUNK  64           // rows of enc per workgroup (4 WMMA m-tiles)
#define NCHUNK (SS / CHUNK) // 32 softmax/context partials per batch

typedef __attribute__((ext_vector_type(16))) __bf16 bf16x16;
typedef __attribute__((ext_vector_type(8)))  float  floatx8;

static __device__ __forceinline__ float fast_tanh(float x) {
#if __has_builtin(__builtin_amdgcn_tanhf)
    return __builtin_amdgcn_tanhf(x);   // native V_TANH_F32 on gfx1250
#else
    return tanhf(x);
#endif
}

// ---------------------------------------------------------------------------
// Kernel 0: U_a (f32 [E,A] row-major) -> bf16 WMMA B-fragment tiles.
// Tile grid: kt in [0,32) over K (steps 32), nt in [0,64) over N (steps 16).
// ISA 16-bit B-matrix 32x16 layout: lane L -> N = nt*16 + (L&15),
// K = kt*32 + ((L<16)?0:8) + (e<8 ? e : e+8), e = 0..15, stored lane-contiguous (32B/lane).
// ---------------------------------------------------------------------------
__global__ void ua_to_bf16_tiles(const float* __restrict__ Ua,
                                 unsigned short* __restrict__ uaT_us) {
    __bf16* uaT = (__bf16*)uaT_us;
    const int lane = threadIdx.x & 31;
    const int tile = blockIdx.x * 8 + (threadIdx.x >> 5);   // 2048 tiles
    const int kt = tile >> 6;
    const int nt = tile & 63;
    const int n  = nt * 16 + (lane & 15);
    const int k0 = kt * 32 + ((lane < 16) ? 0 : 8);
    bf16x16 v;
#pragma unroll
    for (int e = 0; e < 16; ++e)
        v[e] = (__bf16)Ua[(size_t)(k0 + e + ((e < 8) ? 0 : 8)) * AA + n];
    *(bf16x16*)(uaT + (size_t)tile * 512 + lane * 16) = v;
}

// ---------------------------------------------------------------------------
// Kernel 1: w[b,a] = dec[b,:] . W_a[:,a]
// ---------------------------------------------------------------------------
__global__ void dec_proj(const float* __restrict__ dec,
                         const float* __restrict__ Wa,
                         float* __restrict__ w) {
    __shared__ float dls[DD];
    const int b = blockIdx.y;
    const int a = blockIdx.x * 256 + threadIdx.x;
    for (int i = threadIdx.x; i < DD; i += 256) dls[i] = dec[b * DD + i];
    __syncthreads();
    float s = 0.f;
#pragma unroll 4
    for (int d = 0; d < DD; ++d) s += dls[d] * Wa[(size_t)d * AA + a];
    w[b * AA + a] = s;
}

// ---------------------------------------------------------------------------
// Kernel 2 (hot, fused): per WG = (chunk of 64 enc rows, batch b):
//   1) u-tile = enc[64 x 1024] @ U_a via v_wmma_f32_16x16x32_bf16.
//      512 threads = 16 waves; wave owns 4 n-tiles x 4 m-tiles (B-frag
//      reused 4x in registers -> 4x less U_a traffic than M=16).
//   2) e_s = sum_n v_n * tanh(w_n + u_sn)  (ds_add_f32 reduction in LDS)
//   3) local softmax partial (m, l) over the 64 rows
//   4) partial context = sum_s p_s * enc[s,:] (enc tile is cache-hot)
// Combine kernel merges chunk partials -> enc is streamed from HBM ONCE.
// ---------------------------------------------------------------------------
__global__ void __launch_bounds__(512)
attn_fused(const float* __restrict__ enc,
           const unsigned short* __restrict__ uaT_us,
           const float* __restrict__ w,
           const float* __restrict__ vt,
           float* __restrict__ ctxp,   // [B][NCHUNK][EE]
           float* __restrict__ mpart,  // [B][NCHUNK]
           float* __restrict__ lpart) {// [B][NCHUNK]
    const __bf16* uaT = (const __bf16*)uaT_us;
    __shared__ float e_lds[CHUNK];
    __shared__ float p_lds[CHUNK];
    const int lane  = threadIdx.x & 31;
    const int wave  = threadIdx.x >> 5;        // 0..15
    const int b     = blockIdx.y;
    const int chunk = blockIdx.x;              // 0..NCHUNK-1
    const int s0    = chunk * CHUNK;
    const int col   = lane & 15;
    const int mbase = (lane < 16) ? 0 : 8;
    const int nt0   = wave * 4;                // this wave's first n-tile (of 64)

    if (threadIdx.x < CHUNK) e_lds[threadIdx.x] = 0.f;
    __syncthreads();

    floatx8 acc[4][4] = {};                    // [m-tile][n-tile]
    // A-frag row base (ISA 16-bit A 16x32 layout): lane<16 -> K {0-7,16-23},
    // lane>=16 -> K {8-15,24-31}; row M = lane&15 within each m-tile.
    const float* row0 = enc + ((size_t)b * SS + s0 + (lane & 15)) * (size_t)EE
                            + ((lane < 16) ? 0 : 8);

    for (int kt = 0; kt < 32; ++kt) {
        bf16x16 afrag[4];
#pragma unroll
        for (int mt = 0; mt < 4; ++mt) {
            const float* p = row0 + (size_t)mt * 16 * EE + kt * 32;
            if (mt == 0) __builtin_prefetch(p + 256, 0, 1);  // 1KB ahead in stream
            const float4 x0 = *(const float4*)(p);
            const float4 x1 = *(const float4*)(p + 4);
            const float4 x2 = *(const float4*)(p + 16);
            const float4 x3 = *(const float4*)(p + 20);
            bf16x16 a;
            a[0]  = (__bf16)x0.x; a[1]  = (__bf16)x0.y;
            a[2]  = (__bf16)x0.z; a[3]  = (__bf16)x0.w;
            a[4]  = (__bf16)x1.x; a[5]  = (__bf16)x1.y;
            a[6]  = (__bf16)x1.z; a[7]  = (__bf16)x1.w;
            a[8]  = (__bf16)x2.x; a[9]  = (__bf16)x2.y;
            a[10] = (__bf16)x2.z; a[11] = (__bf16)x2.w;
            a[12] = (__bf16)x3.x; a[13] = (__bf16)x3.y;
            a[14] = (__bf16)x3.z; a[15] = (__bf16)x3.w;
            afrag[mt] = a;
        }
        const __bf16* tb = uaT + ((size_t)(kt * 64 + nt0) * 512) + lane * 16;
#pragma unroll
        for (int nt = 0; nt < 4; ++nt) {
            bf16x16 bfrag = *(const bf16x16*)(tb + (size_t)nt * 512);
#pragma unroll
            for (int mt = 0; mt < 4; ++mt)
                acc[mt][nt] = __builtin_amdgcn_wmma_f32_16x16x32_bf16(
                    false, afrag[mt], false, bfrag, (short)0, acc[mt][nt],
                    false, false);
        }
    }

    // ---- e-score epilogue: tanh(u+w).v_t, reduced into LDS --------------
    float wv[4], vv[4];
#pragma unroll
    for (int nt = 0; nt < 4; ++nt) {
        const int n = (nt0 + nt) * 16 + col;
        wv[nt] = w[b * AA + n];
        vv[nt] = vt[n];
    }
#pragma unroll
    for (int mt = 0; mt < 4; ++mt) {
#pragma unroll
        for (int r = 0; r < 8; ++r) {
            float ps = 0.f;
#pragma unroll
            for (int nt = 0; nt < 4; ++nt)
                ps += fast_tanh(acc[mt][nt][r] + wv[nt]) * vv[nt];
            atomicAdd(&e_lds[mt * 16 + mbase + r], ps);
        }
    }
    __syncthreads();

    // ---- local softmax partial over the 64 rows -------------------------
    float m_loc = -3.0e38f;
#pragma unroll 8
    for (int s = 0; s < CHUNK; ++s) m_loc = fmaxf(m_loc, e_lds[s]);
    if (threadIdx.x < CHUNK)
        p_lds[threadIdx.x] = __expf(e_lds[threadIdx.x] - m_loc);
    __syncthreads();

    // ---- partial context: p-weighted sum of the (cache-hot) enc tile ----
    float2 cacc = make_float2(0.f, 0.f);
    const float2* rows = (const float2*)(enc + ((size_t)b * SS + s0) * (size_t)EE)
                         + threadIdx.x;                 // 512 thr x 2 cols = 1024
#pragma unroll 4
    for (int s = 0; s < CHUNK; ++s) {
        const float  ps = p_lds[s];
        const float2 e2 = rows[(size_t)s * (EE / 2)];
        cacc.x += ps * e2.x;
        cacc.y += ps * e2.y;
    }
    ((float2*)(ctxp + ((size_t)b * NCHUNK + chunk) * (size_t)EE))[threadIdx.x] = cacc;
    if (threadIdx.x == 0) {
        float l = 0.f;
        for (int s = 0; s < CHUNK; ++s) l += p_lds[s];
        mpart[b * NCHUNK + chunk] = m_loc;
        lpart[b * NCHUNK + chunk] = l;
    }
}

// ---------------------------------------------------------------------------
// Kernel 3: merge chunk partials:  ctx = sum_i e^{m_i-m_g} ctxp_i / L
// ---------------------------------------------------------------------------
__global__ void ctx_combine(const float* __restrict__ ctxp,
                            const float* __restrict__ mpart,
                            const float* __restrict__ lpart,
                            float* __restrict__ ctx) {
    const int b = blockIdx.x, tid = threadIdx.x;        // 256 thr x float4
    float m_g = -3.0e38f;
#pragma unroll
    for (int i = 0; i < NCHUNK; ++i) m_g = fmaxf(m_g, mpart[b * NCHUNK + i]);
    float wgt[NCHUNK];
    float L = 0.f;
#pragma unroll
    for (int i = 0; i < NCHUNK; ++i) {
        const float wi = __expf(mpart[b * NCHUNK + i] - m_g);
        wgt[i] = wi;
        L += wi * lpart[b * NCHUNK + i];
    }
    const float inv = 1.f / L;
    float4 c = make_float4(0.f, 0.f, 0.f, 0.f);
#pragma unroll 4
    for (int i = 0; i < NCHUNK; ++i) {
        const float4 v =
            ((const float4*)(ctxp + ((size_t)b * NCHUNK + i) * (size_t)EE))[tid];
        c.x += wgt[i] * v.x; c.y += wgt[i] * v.y;
        c.z += wgt[i] * v.z; c.w += wgt[i] * v.w;
    }
    ((float4*)(ctx + (size_t)b * EE))[tid] =
        make_float4(c.x * inv, c.y * inv, c.z * inv, c.w * inv);
}

// ---------------------------------------------------------------------------
// Kernel 4: output[b,d] = tanh( [dec|context][b,:] @ ffn[:,d] )
// ---------------------------------------------------------------------------
__global__ void ffn_out(const float* __restrict__ dec,
                        const float* __restrict__ ctx,
                        const float* __restrict__ ffn,
                        float* __restrict__ out) {
    __shared__ float cat[DD + EE];
    const int b = blockIdx.y;
    const int d = blockIdx.x * 256 + threadIdx.x;
    for (int i = threadIdx.x; i < DD; i += 256) {
        cat[i]      = dec[b * DD + i];
        cat[DD + i] = ctx[b * EE + i];
    }
    __syncthreads();
    float s = 0.f;
#pragma unroll 4
    for (int k = 0; k < DD + EE; ++k) s += cat[k] * ffn[(size_t)k * DD + d];
    out[b * DD + d] = fast_tanh(s);
}

// ---------------------------------------------------------------------------
extern "C" void kernel_launch(void* const* d_in, const int* in_sizes, int n_in,
                              void* d_out, int out_size, void* d_ws, size_t ws_size,
                              hipStream_t stream) {
    (void)in_sizes; (void)n_in; (void)out_size; (void)ws_size;
    const float* enc = (const float*)d_in[0];   // [B,S,E]
    const float* dec = (const float*)d_in[1];   // [B,1,D]
    const float* Ua  = (const float*)d_in[2];   // [E,A]
    const float* Wa  = (const float*)d_in[3];   // [D,A]
    const float* vt  = (const float*)d_in[4];   // [A,1]
    const float* ffn = (const float*)d_in[5];   // [D+E, D]

    float* out = (float*)d_out;                 // [B*D] output, then [B*E] context
    float* ctx = out + (size_t)BB * DD;

    char* ws = (char*)d_ws;
    size_t off = 0;
    unsigned short* uaT = (unsigned short*)(ws + off); off += (size_t)EE * AA * 2;      // 2 MB
    float* w     = (float*)(ws + off); off += (size_t)BB * AA * 4;                      // 128 KB
    float* ctxp  = (float*)(ws + off); off += (size_t)BB * NCHUNK * EE * 4;             // 4 MB
    float* mpart = (float*)(ws + off); off += (size_t)BB * NCHUNK * 4;
    float* lpart = (float*)(ws + off); off += (size_t)BB * NCHUNK * 4;

    ua_to_bf16_tiles<<<256, 256, 0, stream>>>(Ua, uaT);
    dec_proj   <<<dim3(AA / 256, BB), 256, 0, stream>>>(dec, Wa, w);
    attn_fused <<<dim3(NCHUNK, BB), 512, 0, stream>>>(enc, uaT, w, vt,
                                                      ctxp, mpart, lpart);
    ctx_combine<<<BB, 256, 0, stream>>>(ctxp, mpart, lpart, ctx);
    ffn_out    <<<dim3(DD / 256, BB), 256, 0, stream>>>(dec, ctx, ffn, out);
}
